// MultiHeadSelfAttention_9002251453048
// MI455X (gfx1250) — matrix operand order repaired and checked by reading, not compiled, not run on hardware
//
#include <hip/hip_runtime.h>
#include <stdint.h>

typedef __bf16 bf16;
typedef __attribute__((ext_vector_type(16))) __bf16 v16bf;
typedef __attribute__((ext_vector_type(8)))  __bf16 v8bf;
typedef __attribute__((ext_vector_type(4)))  __bf16 v4bf;
typedef __attribute__((ext_vector_type(2)))  __bf16 v2bf;
typedef __attribute__((ext_vector_type(8)))  float  v8f;

#define SEQ_N    4096
#define BATCH    4
#define HEADS    4
#define HDIM     64
#define EMB      256
#define QK_SCALE 0.125f   // 64^-0.5
#define NT       (SEQ_N / 64)

// ---------- helpers ----------
__device__ __forceinline__ v8f vzero8() {
  v8f z;
  #pragma unroll
  for (int i = 0; i < 8; ++i) z[i] = 0.0f;
  return z;
}

// Async global->LDS copy (gfx1250, ASYNCcnt-tracked). lds_off = low 32 bits of
// the flat shared address.
__device__ __forceinline__ void async_ld_b128(unsigned lds_off, unsigned long long gaddr) {
  asm volatile("global_load_async_to_lds_b128 %0, %1, off"
               :: "v"(lds_off), "v"(gaddr) : "memory");
}
__device__ __forceinline__ void wait_asynccnt0() {
  asm volatile("s_wait_asynccnt 0x0" ::: "memory");
}

// A-fragment (16x32 bf16), source row-major, leading dim ld (elements).
__device__ __forceinline__ v16bf load_frag_a(const bf16* base, int ld) {
  int lane = threadIdx.x & 31;
  int r  = lane & 15;
  int hl = lane >> 4;
  const bf16* p = base + r * ld + hl * 8;
  v8bf lo = *reinterpret_cast<const v8bf*>(p);        // K = 8*hl + 0..7
  v8bf hi = *reinterpret_cast<const v8bf*>(p + 16);   // K = 16 + 8*hl + 0..7
  v16bf a;
  #pragma unroll
  for (int i = 0; i < 8; ++i) { a[i] = lo[i]; a[i + 8] = hi[i]; }
  return a;
}

// B-fragment (32x16 bf16): B[k][n], source stored row-major as [n][k].
__device__ __forceinline__ v16bf load_frag_b(const bf16* base, int ld) {
  int lane = threadIdx.x & 31;
  int n  = lane & 15;
  int kh = (lane >> 4) << 3;
  const bf16* p = base + n * ld + kh;
  v8bf lo = *reinterpret_cast<const v8bf*>(p);
  v8bf hi = *reinterpret_cast<const v8bf*>(p + 16);
  v16bf b;
  #pragma unroll
  for (int i = 0; i < 8; ++i) { b[i] = lo[i]; b[i + 8] = hi[i]; }
  return b;
}

__device__ __forceinline__ v8f wmma_bf16(v16bf a, v16bf b, v8f c) {
  return __builtin_amdgcn_wmma_f32_16x16x32_bf16(false, a, false, b,
                                                 (short)0, c, false, false);
}

// ---------- kernel 0: fp32 -> bf16 (native cvt) ----------
__global__ void mhsa_cvt_bf16(const float* __restrict__ src,
                              bf16* __restrict__ dst, int n) {
  int i = blockIdx.x * blockDim.x + threadIdx.x;
  if (i < n) dst[i] = (bf16)src[i];
}

// ---------- kernel 1: QKV projection ----------
__global__ __launch_bounds__(256) void mhsa_qkv_gemm(
    const float* __restrict__ x, const bf16* __restrict__ wq,
    bf16* __restrict__ Q, bf16* __restrict__ K, bf16* __restrict__ V) {
  __shared__ bf16 As[128 * 40];
  __shared__ bf16 Bs[64 * 40];
  int tid = threadIdx.x, wid = tid >> 5, lane = tid & 31;
  int m0 = blockIdx.x * 128;
  int n0 = blockIdx.y * 64;
  int wm = (wid >> 1) * 32;
  int wn = (wid & 1) * 32;

  v8f acc[2][2];
  #pragma unroll
  for (int i = 0; i < 2; ++i)
    #pragma unroll
    for (int j = 0; j < 2; ++j) acc[i][j] = vzero8();

  int arow = tid >> 3, acolg = (tid & 7) * 4;
  int brow = tid >> 2, bcolg = (tid & 3) * 8;
  float4 fx[4];
  v8bf bw;

  auto load_regs = [&](int kt) {
    #pragma unroll
    for (int it = 0; it < 4; ++it)
      fx[it] = *reinterpret_cast<const float4*>(
          &x[(size_t)(m0 + arow + 32 * it) * EMB + kt + acolg]);
    bw = *reinterpret_cast<const v8bf*>(&wq[(size_t)(n0 + brow) * EMB + kt + bcolg]);
  };
  auto store_lds = [&]() {
    #pragma unroll
    for (int it = 0; it < 4; ++it) {
      v4bf t;
      t[0] = (bf16)fx[it].x; t[1] = (bf16)fx[it].y;
      t[2] = (bf16)fx[it].z; t[3] = (bf16)fx[it].w;
      *reinterpret_cast<v4bf*>(&As[(arow + 32 * it) * 40 + acolg]) = t;
    }
    *reinterpret_cast<v8bf*>(&Bs[brow * 40 + bcolg]) = bw;
  };

  load_regs(0);
  for (int it2 = 0; it2 < 8; ++it2) {
    store_lds();
    __syncthreads();
    if (it2 + 1 < 8) load_regs((it2 + 1) * 32);   // overlap with WMMA
    #pragma unroll
    for (int sm = 0; sm < 2; ++sm) {
      v16bf a = load_frag_a(&As[(wm + sm * 16) * 40], 40);
      #pragma unroll
      for (int sn = 0; sn < 2; ++sn) {
        v16bf b = load_frag_b(&Bs[(wn + sn * 16) * 40], 40);
        acc[sm][sn] = wmma_bf16(a, b, acc[sm][sn]);
      }
    }
    __syncthreads();
  }

  int hl = lane >> 4, nn = lane & 15;
  #pragma unroll
  for (int sm = 0; sm < 2; ++sm)
    #pragma unroll
    for (int sn = 0; sn < 2; ++sn) {
      int gc = n0 + wn + sn * 16 + nn;
      int s = gc >> 8, h = (gc >> 6) & 3, d = gc & 63;
      bf16* dst = (s == 0) ? Q : (s == 1) ? K : V;
      float sc = (s == 0) ? QK_SCALE : 1.0f;
      #pragma unroll
      for (int i = 0; i < 8; ++i) {
        int gm = m0 + wm + sm * 16 + i + 8 * hl;
        int b  = gm >> 12, n = gm & 4095;
        dst[(((size_t)(b * HEADS + h)) * SEQ_N + n) * HDIM + d] =
            (bf16)(acc[sm][sn][i] * sc);
      }
    }
}

// ---------- kernel 2: flash attention ----------
// grid (N/128, B*H). 8 waves; wave w owns rows 16w..16w+15.
// K tiles: async global->LDS ping-pong. V tiles: reg->LDS transposed ping-pong.
// One barrier per j-tile. Q buffer is recycled as the P staging buffer.
__global__ __launch_bounds__(256) void mhsa_flash_attn(
    const bf16* __restrict__ Q, const bf16* __restrict__ K,
    const bf16* __restrict__ V, bf16* __restrict__ AO) {
  __shared__ bf16 QPs[128 * 72];      // Q tile, later per-wave P staging
  __shared__ bf16 Ks[2][64 * 72];     // ping-pong K (async fill)
  __shared__ bf16 Vt[2][64 * 72];     // ping-pong V^T [d][key]
  int tid = threadIdx.x, wid = tid >> 5, lane = tid & 31;
  int hl = lane >> 4, nn = lane & 15;
  int bh = blockIdx.y;
  int row0 = blockIdx.x * 128;
  const bf16* Qb = Q + (size_t)bh * SEQ_N * HDIM;
  const bf16* Kb = K + (size_t)bh * SEQ_N * HDIM;
  const bf16* Vb = V + (size_t)bh * SEQ_N * HDIM;

  { // load Q tile 128x64
    int r = tid >> 1, cg = (tid & 1) * 32;
    #pragma unroll
    for (int j = 0; j < 4; ++j) {
      v8bf v = *reinterpret_cast<const v8bf*>(&Qb[(size_t)(row0 + r) * HDIM + cg + j * 8]);
      *reinterpret_cast<v8bf*>(&QPs[r * 72 + cg + j * 8]) = v;
    }
  }
  __syncthreads();
  v16bf qa0 = load_frag_a(&QPs[(wid * 16) * 72], 72);
  v16bf qa1 = load_frag_a(&QPs[(wid * 16) * 72 + 32], 72);

  v8f o[4];
  float mstate[8], lstate[8];
  #pragma unroll
  for (int i = 0; i < 4; ++i) o[i] = vzero8();
  #pragma unroll
  for (int i = 0; i < 8; ++i) { mstate[i] = -1e30f; lstate[i] = 0.0f; }

  // K: thread covers row kr, cols kc..kc+15.  V: rows vr,vr+1, cols vc..vc+7.
  int kr = tid >> 2, kc = (tid & 3) * 16;
  int vr = (tid >> 3) * 2, vc = (tid & 7) * 8;
  v8bf vreg0, vreg1;

  auto async_k = [&](int j0, int buf) {
    unsigned l0 = (unsigned)(size_t)&Ks[buf][kr * 72 + kc];
    unsigned long long g = (unsigned long long)(size_t)&Kb[(size_t)(j0 + kr) * HDIM + kc];
    async_ld_b128(l0, g);
    async_ld_b128(l0 + 16u, g + 16ull);
  };
  auto load_v = [&](int j0) {
    vreg0 = *reinterpret_cast<const v8bf*>(&Vb[(size_t)(j0 + vr) * HDIM + vc]);
    vreg1 = *reinterpret_cast<const v8bf*>(&Vb[(size_t)(j0 + vr + 1) * HDIM + vc]);
  };
  auto store_vt = [&](int buf) {
    #pragma unroll
    for (int e = 0; e < 8; ++e) {   // transpose 2-key x 8-d chunk
      v2bf t; t[0] = vreg0[e]; t[1] = vreg1[e];
      *reinterpret_cast<v2bf*>(&Vt[buf][(vc + e) * 72 + vr]) = t;
    }
  };

  async_k(0, 0);
  load_v(0);
  store_vt(0);
  wait_asynccnt0();
  __syncthreads();

  for (int jt = 0; jt < NT; ++jt) {
    int cur = jt & 1, nxt = cur ^ 1;
    if (jt + 1 < NT) {                 // fill other buffers during compute
      async_k((jt + 1) * 64, nxt);
      load_v((jt + 1) * 64);
    }
    if (jt + 2 < NT) {                 // pre-warm L2 / WGP$
      __builtin_prefetch(&Kb[(size_t)((jt + 2) * 64 + kr) * HDIM + kc], 0, 1);
      __builtin_prefetch(&Vb[(size_t)((jt + 2) * 64 + vr) * HDIM + vc], 0, 1);
    }

    // S = Q Kt  (Q pre-scaled by QK_SCALE)
    v8f s[4];
    #pragma unroll
    for (int sub = 0; sub < 4; ++sub) {
      v16bf b0 = load_frag_b(&Ks[cur][(sub * 16) * 72], 72);
      v16bf b1 = load_frag_b(&Ks[cur][(sub * 16) * 72 + 32], 72);
      v8f c = vzero8();
      c = wmma_bf16(qa0, b0, c);
      c = wmma_bf16(qa1, b1, c);
      s[sub] = c;
    }

    // online softmax
    #pragma unroll
    for (int i = 0; i < 8; ++i) {
      float v = fmaxf(fmaxf(s[0][i], s[1][i]), fmaxf(s[2][i], s[3][i]));
      v = fmaxf(v, __shfl_xor(v, 1, 32));
      v = fmaxf(v, __shfl_xor(v, 2, 32));
      v = fmaxf(v, __shfl_xor(v, 4, 32));
      v = fmaxf(v, __shfl_xor(v, 8, 32));
      float mi = fmaxf(mstate[i], v);
      float sc = __expf(mstate[i] - mi);
      float p0 = __expf(s[0][i] - mi), p1 = __expf(s[1][i] - mi);
      float p2 = __expf(s[2][i] - mi), p3 = __expf(s[3][i] - mi);
      s[0][i] = p0; s[1][i] = p1; s[2][i] = p2; s[3][i] = p3;
      float rs = p0 + p1 + p2 + p3;
      rs += __shfl_xor(rs, 1, 32);
      rs += __shfl_xor(rs, 2, 32);
      rs += __shfl_xor(rs, 4, 32);
      rs += __shfl_xor(rs, 8, 32);
      lstate[i] = lstate[i] * sc + rs;
      mstate[i] = mi;
      o[0][i] *= sc; o[1][i] *= sc; o[2][i] *= sc; o[3][i] *= sc;
    }

    // stage P (C-layout -> row-major bf16) in recycled Q buffer (own region)
    bf16* myP = &QPs[wid * 16 * 72];
    #pragma unroll
    for (int sub = 0; sub < 4; ++sub)
      #pragma unroll
      for (int i = 0; i < 8; ++i)
        myP[(i + 8 * hl) * 72 + sub * 16 + nn] = (bf16)s[sub][i];

    // O += P @ V
    v16bf pa0 = load_frag_a(myP, 72);
    v16bf pa1 = load_frag_a(myP + 32, 72);
    #pragma unroll
    for (int sub = 0; sub < 4; ++sub) {
      v16bf b0 = load_frag_b(&Vt[cur][(sub * 16) * 72], 72);
      v16bf b1 = load_frag_b(&Vt[cur][(sub * 16) * 72 + 32], 72);
      o[sub] = wmma_bf16(pa0, b0, o[sub]);
      o[sub] = wmma_bf16(pa1, b1, o[sub]);
    }

    if (jt + 1 < NT) store_vt(nxt);    // Vt[nxt] dead since previous barrier
    wait_asynccnt0();                  // K[nxt] fill complete (this wave)
    __syncthreads();                   // single barrier per tile
  }

  // write attn output: AO[b, seq, h*64+d] bf16
  int b = bh >> 2, h = bh & 3;
  #pragma unroll
  for (int sub = 0; sub < 4; ++sub)
    #pragma unroll
    for (int i = 0; i < 8; ++i) {
      int seq = row0 + wid * 16 + i + 8 * hl;
      int d   = sub * 16 + nn;
      AO[((size_t)b * SEQ_N + seq) * EMB + h * HDIM + d] =
          (bf16)(o[sub][i] / lstate[i]);
    }
}

// ---------- kernel 3: output projection + bias ----------
__global__ __launch_bounds__(256) void mhsa_proj_gemm(
    const bf16* __restrict__ A, const bf16* __restrict__ wp,
    const float* __restrict__ bias, float* __restrict__ out) {
  __shared__ bf16 As[128 * 40];
  __shared__ bf16 Bs[64 * 40];
  int tid = threadIdx.x, wid = tid >> 5, lane = tid & 31;
  int m0 = blockIdx.x * 128;
  int n0 = blockIdx.y * 64;
  int wm = (wid >> 1) * 32;
  int wn = (wid & 1) * 32;

  v8f acc[2][2];
  #pragma unroll
  for (int i = 0; i < 2; ++i)
    #pragma unroll
    for (int j = 0; j < 2; ++j) acc[i][j] = vzero8();

  int arow = tid >> 1, acg = (tid & 1) * 16;
  int brow = tid >> 2, bcolg = (tid & 3) * 8;
  v8bf aw0, aw1, bw;

  auto load_regs = [&](int kt) {
    aw0 = *reinterpret_cast<const v8bf*>(&A[(size_t)(m0 + arow) * EMB + kt + acg]);
    aw1 = *reinterpret_cast<const v8bf*>(&A[(size_t)(m0 + arow) * EMB + kt + acg + 8]);
    bw  = *reinterpret_cast<const v8bf*>(&wp[(size_t)(n0 + brow) * EMB + kt + bcolg]);
  };
  auto store_lds = [&]() {
    *reinterpret_cast<v8bf*>(&As[arow * 40 + acg]) = aw0;
    *reinterpret_cast<v8bf*>(&As[arow * 40 + acg + 8]) = aw1;
    *reinterpret_cast<v8bf*>(&Bs[brow * 40 + bcolg]) = bw;
  };

  load_regs(0);
  for (int it2 = 0; it2 < 8; ++it2) {
    store_lds();
    __syncthreads();
    if (it2 + 1 < 8) load_regs((it2 + 1) * 32);
    #pragma unroll
    for (int sm = 0; sm < 2; ++sm) {
      v16bf a = load_frag_a(&As[(wm + sm * 16) * 40], 40);
      #pragma unroll
      for (int sn = 0; sn < 2; ++sn) {
        v16bf b = load_frag_b(&Bs[(wn + sn * 16) * 40], 40);
        acc[sm][sn] = wmma_bf16(a, b, acc[sm][sn]);
      }
    }
    __syncthreads();
  }

  int hl = lane >> 4, nn = lane & 15;
  #pragma unroll
  for (int sm = 0; sm < 2; ++sm)
    #pragma unroll
    for (int sn = 0; sn < 2; ++sn) {
      int gc = n0 + wn + sn * 16 + nn;
      float bv = bias[gc];
      #pragma unroll
      for (int i = 0; i < 8; ++i) {
        int gm = m0 + wm + sm * 16 + i + 8 * hl;
        out[(size_t)gm * EMB + gc] = acc[sm][sn][i] + bv;
      }
    }
}

// ---------- launcher ----------
extern "C" void kernel_launch(void* const* d_in, const int* in_sizes, int n_in,
                              void* d_out, int out_size, void* d_ws, size_t ws_size,
                              hipStream_t stream) {
  const float* x      = (const float*)d_in[0];
  const float* w_qkv  = (const float*)d_in[1];
  const float* w_proj = (const float*)d_in[2];
  const float* b_proj = (const float*)d_in[3];
  float* out = (float*)d_out;

  bf16* wq_bf = (bf16*)d_ws;                    // 768*256
  bf16* wp_bf = wq_bf + 768 * 256;              // 256*256
  bf16* Q     = wp_bf + 256 * 256;              // 16*4096*64 each
  bf16* Kb    = Q  + (size_t)16 * SEQ_N * HDIM;
  bf16* Vb    = Kb + (size_t)16 * SEQ_N * HDIM;
  bf16* AO    = Vb + (size_t)16 * SEQ_N * HDIM; // 16384*256

  mhsa_cvt_bf16<<<(768 * 256 + 255) / 256, 256, 0, stream>>>(w_qkv, wq_bf, 768 * 256);
  mhsa_cvt_bf16<<<(256 * 256 + 255) / 256, 256, 0, stream>>>(w_proj, wp_bf, 256 * 256);
  mhsa_qkv_gemm<<<dim3(128, 12), 256, 0, stream>>>(x, wq_bf, Q, Kb, Vb);
  mhsa_flash_attn<<<dim3(32, 16), 256, 0, stream>>>(Q, Kb, Vb, AO);
  mhsa_proj_gemm<<<dim3(128, 4), 256, 0, stream>>>(AO, wp_bf, b_proj, out);
}
